// Backprop_29188597744223
// MI455X (gfx1250) — matrix operand order repaired and checked by reading, not compiled, not run on hardware
//
#include <hip/hip_runtime.h>
#include <hip/hip_bf16.h>

typedef __attribute__((ext_vector_type(16))) _Float16 v16h;
typedef __attribute__((ext_vector_type(8)))  _Float16 v8h;
typedef __attribute__((ext_vector_type(8)))  float    v8f;

#define LRC   0.01f
#define EPSN  1e-8f
#define DD    1024
#define HH    256
#define OO    64
#define NSTEP 1024
#define NT    1024   // threads per block (32 wave32s)

// ---------------------------------------------------------------------------
// Swizzled f16 shadow-weight index: element (d=k-row, j=col) of a K x (16*nTiles)
// matrix is stored exactly where the WMMA B operand wants it:
//   block = (d/32)*nTiles + (j/16), slot lane = (j&15) | ((d>>4)&1)<<4,
//   elem  = d&15  -> halves index = (block*32 + lane)*16 + elem.
// Each lane loads its full 16-half B fragment as ONE contiguous 32B chunk.
// ---------------------------------------------------------------------------
__device__ inline size_t swzIdx(int d, int j, int nTiles) {
  size_t block = (size_t)(d >> 5) * nTiles + (j >> 4);
  int    lane  = (j & 15) | (((d >> 4) & 1) << 4);
  return ((block * 32 + lane) << 4) + (d & 15);
}

// GEMV tile via WMMA: z[col0+n] = sum_k act[k] * W[k, col0+n], n = 0..15.
// A broadcasts the f16 activation vector across the 16 M rows (this file's k
// order k = 16*(lane/16) + s, same as B): per lane the fragment is two contiguous 16B LDS chunks.
// B comes from the swizzled f16 shadow (one v16h global load per lane/K-block).
// Returns the tile's z value for column tile*16 + (lane&15) in every lane.
__device__ inline float gemv16(const _Float16* __restrict__ sh,
                               const _Float16* __restrict__ actH,
                               int kb0, int nkb, int tile, int nTiles, int lane) {
  v8f acc0 = {}, acc1 = {};
  const int ha = (lane & 16) ? 16 : 0;
  const _Float16* base =
      sh + ((((size_t)kb0 * nTiles + tile) * 32 + lane) << 4);
  const size_t kstride = (size_t)nTiles * 512;  // halves per K-block of 32
  for (int kb = 0; kb < nkb; kb += 2) {
    const int k = (kb0 + kb) * 32;
    v8h lo0 = *(const v8h*)(actH + k + ha);
    v8h hi0 = *(const v8h*)(actH + k + 8 + ha);
    v8h lo1 = *(const v8h*)(actH + k + 32 + ha);
    v8h hi1 = *(const v8h*)(actH + k + 40 + ha);
    v16h A0 = __builtin_shufflevector(lo0, hi0, 0, 1, 2, 3, 4, 5, 6, 7,
                                      8, 9, 10, 11, 12, 13, 14, 15);
    v16h A1 = __builtin_shufflevector(lo1, hi1, 0, 1, 2, 3, 4, 5, 6, 7,
                                      8, 9, 10, 11, 12, 13, 14, 15);
    v16h B0 = *(const v16h*)(base + (size_t)kb * kstride);
    v16h B1 = *(const v16h*)(base + (size_t)(kb + 1) * kstride);
    acc0 = __builtin_amdgcn_wmma_f32_16x16x32_f16(false, A0, false, B0,
                                                  (short)0, acc0, false, false);
    acc1 = __builtin_amdgcn_wmma_f32_16x16x32_f16(false, A1, false, B1,
                                                  (short)0, acc1, false, false);
  }
  return acc0[0] + acc1[0];
}

__device__ inline float wred(float v) {
#pragma unroll
  for (int m = 16; m > 0; m >>= 1) v += __shfl_xor(v, m, 32);
  return v;
}

__device__ inline float fast_sigmoid(float z) {
  return __builtin_amdgcn_rcpf(1.0f + __expf(-z));
}

__global__ __launch_bounds__(NT)
void Backprop_29188597744223_kernel(const float* __restrict__ xg,
                                    const float* __restrict__ tg,
                                    const float* __restrict__ w1i,
                                    const float* __restrict__ b1i,
                                    const float* __restrict__ w2i,
                                    const float* __restrict__ b2i,
                                    const float* __restrict__ w3i,
                                    const float* __restrict__ b3i,
                                    float* __restrict__ U1,
                                    float* __restrict__ U2,
                                    float* __restrict__ U3,
                                    _Float16* __restrict__ sh1,
                                    _Float16* __restrict__ sh2,
                                    _Float16* __restrict__ sh3) {
  __shared__ float xs[DD];
  __shared__ __align__(32) _Float16 xh[DD];
  __shared__ __align__(32) _Float16 a1h[HH];
  __shared__ __align__(32) _Float16 a2h[HH];
  __shared__ float redA[NT];
  __shared__ float redB[32];
  __shared__ float a1[HH], a2[HH], a3[OO];
  __shared__ float ttl[OO], d3v[OO], smx[OO];
  __shared__ float o1v[HH], o2v[HH];
  __shared__ float c1[HH], c2[HH], c3[OO];
  __shared__ float s1s, s2s, s3s, r1s, r2s, r3s, invS;

  const int tid  = threadIdx.x;
  const int w    = tid >> 5;
  const int lane = tid & 31;

  // ---- init: f32 masters into d_out, f16 swizzled shadows into d_ws
  for (int e = tid; e < DD * HH; e += NT) {
    float v = w1i[e];
    U1[e] = v;
    sh1[swzIdx(e >> 8, e & 255, 16)] = (_Float16)v;
  }
  for (int e = tid; e < HH * HH; e += NT) {
    float v = w2i[e];
    U2[e] = v;
    sh2[swzIdx(e >> 8, e & 255, 16)] = (_Float16)v;
  }
  for (int e = tid; e < HH * OO; e += NT) {
    float v = w3i[e];
    U3[e] = v;
    sh3[swzIdx(e >> 6, e & 63, 4)] = (_Float16)v;
  }
  if (tid < HH) { c1[tid] = b1i[tid]; c2[tid] = b2i[tid]; }
  if (tid < OO) c3[tid] = b3i[tid];
  if (tid == 0) { s1s = s2s = s3s = r1s = r2s = r3s = 1.0f; }
  __syncthreads();

  for (int t = 0; t < NSTEP; ++t) {
    // ---- async-copy sample row into LDS (f32 master) + f16 copy for WMMA A
    {
      unsigned ldsoff = (unsigned)(uintptr_t)(&xs[tid]);
      const float* gp = xg + (size_t)t * DD + tid;
      asm volatile("global_load_async_to_lds_b32 %0, %1, off"
                   :: "v"(ldsoff), "v"(gp) : "memory");
      xh[tid] = (_Float16)(*gp);  // same line, L0/L2 hit
      asm volatile("s_wait_asynccnt 0" ::: "memory");
    }
    if (tid < OO) ttl[tid] = tg[(size_t)t * OO + tid];
    if (t + 1 < NSTEP) __builtin_prefetch(&xg[(size_t)(t + 1) * DD + tid], 0, 1);
    __syncthreads();

    // ---- layer 1: z1 = x @ (s1*W1) + r1*b1 ; 16 col-tiles x 2 K-halves
    {
      int tile = w >> 1, hk = w & 1;
      float z = gemv16(sh1, xh, hk * 16, 16, tile, 16, lane);
      if (lane < 16) redA[w * 16 + lane] = z;
    }
    __syncthreads();
    if (tid < HH) {
      float z = redA[(tid >> 4) * 32 + (tid & 15)] +
                redA[(tid >> 4) * 32 + 16 + (tid & 15)];
      float s = fast_sigmoid(s1s * z + r1s * c1[tid]);
      a1[tid] = s;
      a1h[tid] = (_Float16)s;
    }
    __syncthreads();

    // ---- layer 2
    {
      int tile = w >> 1, hk = w & 1;
      float z = gemv16(sh2, a1h, hk * 4, 4, tile, 16, lane);
      if (lane < 16) redA[w * 16 + lane] = z;
    }
    __syncthreads();
    if (tid < HH) {
      float z = redA[(tid >> 4) * 32 + (tid & 15)] +
                redA[(tid >> 4) * 32 + 16 + (tid & 15)];
      float s = fast_sigmoid(s2s * z + r2s * c2[tid]);
      a2[tid] = s;
      a2h[tid] = (_Float16)s;
    }
    __syncthreads();

    // ---- layer 3 (4 col-tiles -> waves 0..7)
    if (w < 8) {
      int tile = w >> 1, hk = w & 1;
      float z = gemv16(sh3, a2h, hk * 4, 4, tile, 4, lane);
      if (lane < 16) redA[w * 16 + lane] = z;
    }
    __syncthreads();
    if (tid < OO) {
      float z = redA[(tid >> 4) * 32 + (tid & 15)] +
                redA[(tid >> 4) * 32 + 16 + (tid & 15)];
      a3[tid] = fast_sigmoid(s3s * z + r3s * c3[tid]);
    }
    __syncthreads();

    // ---- softmax(-a3) and d3
    if (tid == 0) {
      float m = -3.4e38f;
      for (int k = 0; k < OO; ++k) m = fmaxf(m, -a3[k]);
      float ssum = 0.0f;
      for (int k = 0; k < OO; ++k) {
        float e = __expf(-a3[k] - m);
        smx[k] = e; ssum += e;
      }
      invS = __builtin_amdgcn_rcpf(ssum);
    }
    __syncthreads();
    if (tid < OO) {
      float out = smx[tid] * invS;
      d3v[tid] = (ttl[tid] - out) * a3[tid] * (1.0f - a3[tid]);
    }
    __syncthreads();

    // ---- o2 = (s3 * W3 @ d3) * a2*(1-a2)   (256 rows, 4 threads/row)
    {
      int i = tid >> 2, p = tid & 3;
      const float* row = U3 + (size_t)i * OO + p * 16;
      float s = 0.0f;
#pragma unroll
      for (int k = 0; k < 16; ++k) s += row[k] * d3v[p * 16 + k];
      redA[tid] = s;
    }
    __syncthreads();
    if (tid < HH) {
      float s = redA[4 * tid] + redA[4 * tid + 1] + redA[4 * tid + 2] + redA[4 * tid + 3];
      o2v[tid] = s3s * s * a2[tid] * (1.0f - a2[tid]);
    }
    __syncthreads();

    // ---- o1 = (s2 * W2 @ o2) * a1*(1-a1)
    {
      int i = tid >> 2, p = tid & 3;
      const float* row = U2 + (size_t)i * HH + p * 64;
      float s = 0.0f;
      for (int k = 0; k < 64; ++k) s += row[k] * o2v[p * 64 + k];
      redA[tid] = s;
    }
    __syncthreads();
    if (tid < HH) {
      float s = redA[4 * tid] + redA[4 * tid + 1] + redA[4 * tid + 2] + redA[4 * tid + 3];
      o1v[tid] = s2s * s * a1[tid] * (1.0f - a1[tid]);
    }
    __syncthreads();

    // ---- layer-3 update + fused Frobenius norm (lazy scale), shadow refresh
    {
      float wl = 0.0f, bl = 0.0f;
      const float s3 = s3s;
      for (int e = tid; e < HH * OO; e += NT) {
        int i = e >> 6, k = e & 63;
        float nv = s3 * U3[e] - LRC * a2[i] * d3v[k];
        U3[e] = nv;
        sh3[swzIdx(i, k, 4)] = (_Float16)nv;
        wl += nv * nv;
      }
      if (tid < OO) {
        float nc = r3s * c3[tid] - LRC * d3v[tid];
        c3[tid] = nc; bl = nc * nc;
      }
      wl = wred(wl); bl = wred(bl);
      if (lane == 0) { redA[w] = wl; redB[w] = bl; }
      __syncthreads();
      if (w == 0) {
        float x = wred(redA[lane]);
        float y = wred(redB[lane]);
        if (lane == 0) {
          s3s = 1.0f / fmaxf(sqrtf(x), EPSN);
          r3s = 1.0f / fmaxf(sqrtf(y), EPSN);
        }
      }
    }
    __syncthreads();

    // ---- layer-2 update + fused norm, shadow refresh
    {
      float wl = 0.0f, bl = 0.0f;
      const float s2 = s2s;
      for (int e = tid; e < HH * HH; e += NT) {
        int i = e >> 8, j = e & 255;
        float nv = s2 * U2[e] - LRC * a1[i] * o2v[j];
        U2[e] = nv;
        sh2[swzIdx(i, j, 16)] = (_Float16)nv;
        wl += nv * nv;
      }
      if (tid < HH) {
        float nc = r2s * c2[tid] - LRC * o2v[tid];
        c2[tid] = nc; bl = nc * nc;
      }
      wl = wred(wl); bl = wred(bl);
      if (lane == 0) { redA[w] = wl; redB[w] = bl; }
      __syncthreads();
      if (w == 0) {
        float x = wred(redA[lane]);
        float y = wred(redB[lane]);
        if (lane == 0) {
          s2s = 1.0f / fmaxf(sqrtf(x), EPSN);
          r2s = 1.0f / fmaxf(sqrtf(y), EPSN);
        }
      }
    }
    __syncthreads();

    // ---- layer-1 update + fused norm, shadow refresh
    {
      float wl = 0.0f, bl = 0.0f;
      const float s1 = s1s;
      for (int e = tid; e < DD * HH; e += NT) {
        int d = e >> 8, j = e & 255;
        float nv = s1 * U1[e] - LRC * xs[d] * o1v[j];
        U1[e] = nv;
        sh1[swzIdx(d, j, 16)] = (_Float16)nv;
        wl += nv * nv;
      }
      if (tid < HH) {
        float nc = r1s * c1[tid] - LRC * o1v[tid];
        c1[tid] = nc; bl = nc * nc;
      }
      wl = wred(wl); bl = wred(bl);
      if (lane == 0) { redA[w] = wl; redB[w] = bl; }
      __syncthreads();
      if (w == 0) {
        float x = wred(redA[lane]);
        float y = wred(redB[lane]);
        if (lane == 0) {
          s1s = 1.0f / fmaxf(sqrtf(x), EPSN);
          r1s = 1.0f / fmaxf(sqrtf(y), EPSN);
        }
      }
    }
    __syncthreads();
  }

  // ---- materialize true weights: W_k = s_k * U_k (d_out already holds U's)
  __syncthreads();
  const float f1 = s1s, f2 = s2s, f3 = s3s;
  for (int e = tid; e < DD * HH; e += NT) U1[e] *= f1;
  for (int e = tid; e < HH * HH; e += NT) U2[e] *= f2;
  for (int e = tid; e < HH * OO; e += NT) U3[e] *= f3;
}

extern "C" void kernel_launch(void* const* d_in, const int* in_sizes, int n_in,
                              void* d_out, int out_size, void* d_ws, size_t ws_size,
                              hipStream_t stream) {
  (void)in_sizes; (void)n_in; (void)out_size; (void)ws_size;
  const float* xg  = (const float*)d_in[0];
  const float* tg  = (const float*)d_in[1];
  const float* w1i = (const float*)d_in[2];
  const float* b1i = (const float*)d_in[3];
  const float* w2i = (const float*)d_in[4];
  const float* b2i = (const float*)d_in[5];
  const float* w3i = (const float*)d_in[6];
  const float* b3i = (const float*)d_in[7];

  float* out = (float*)d_out;               // layout: W1 | W2 | W3
  float* U1 = out;
  float* U2 = out + DD * HH;
  float* U3 = out + DD * HH + HH * HH;

  _Float16* sh1 = (_Float16*)d_ws;          // swizzled f16 shadows
  _Float16* sh2 = sh1 + (size_t)DD * HH;
  _Float16* sh3 = sh2 + (size_t)HH * HH;

  hipLaunchKernelGGL(Backprop_29188597744223_kernel, dim3(1), dim3(NT), 0, stream,
                     xg, tg, w1i, b1i, w2i, b2i, w3i, b3i,
                     U1, U2, U3, sh1, sh2, sh3);
}
